// GatedDeltaNet_87763361726639
// MI455X (gfx1250) — matrix operand order repaired and checked by reading, compiled, not run on hardware
//
#include <hip/hip_runtime.h>
#include <hip/hip_bf16.h>

typedef __attribute__((ext_vector_type(16))) _Float16 v16h;
typedef __attribute__((ext_vector_type(8)))  _Float16 v8h;
typedef __attribute__((ext_vector_type(8)))  float    v8f;

#define B_   2
#define L_   1024
#define D_   1024
#define H_   6
#define DK_  128
#define DV_  256
#define KD_  768     // H*DK
#define VD_  1536    // H*DV
#define KS_  4
#define ROWS (B_*L_) // 2048

// ---------------------------------------------------------------------------
// fp32 -> f16 conversion (row-major, feeds WMMA A operands)
// ---------------------------------------------------------------------------
__global__ void gdn_cvt_f16(const float* __restrict__ in,
                            _Float16* __restrict__ out, int n) {
  int i = blockIdx.x * blockDim.x + threadIdx.x;
  if (i < n) out[i] = (_Float16)in[i];
}

// ---------------------------------------------------------------------------
// fp32 [K,N] -> f16 transposed [N,K]. Weights are one-time converted and
// stay L2-resident (192MB L2 >> 13MB of weights), so B fragments become
// contiguous 32B global loads in the GEMM (no LDS transpose staging at all).
// ---------------------------------------------------------------------------
__global__ void gdn_cvt_f16_t(const float* __restrict__ in,
                              _Float16* __restrict__ out, int K, int N) {
  int idx = blockIdx.x * blockDim.x + threadIdx.x;  // idx = n*K + k
  if (idx >= K * N) return;
  int kx = idx % K;
  int n  = idx / K;
  out[idx] = (_Float16)in[(size_t)kx * N + n];
}

// ---------------------------------------------------------------------------
// LDS-free WMMA GEMM: C[M,N] = A[M,K] * B[K,N] with B supplied transposed
// (BT row-major [N,K], f16). Block = 256 threads (8 waves), block tile
// 64(M) x 128(N); each wave owns a 32x32 tile -> 2 A frags x 2 B frags and
// 4 WMMAs per 32-deep k-step with full operand reuse. No barriers, no DS.
// A fragment: lanes 0-15 K=0-15, lanes 16-31 K=16-31 (two v8h).
// B fragment: per-lane column strip of 16 consecutive K = one aligned v16h.
// Requires M%64==0, N%128==0, K%32==0 (all shapes comply).
// ---------------------------------------------------------------------------
__global__ __launch_bounds__(256) void gdn_gemm_f16_wmma_nt(
    const _Float16* __restrict__ A, const _Float16* __restrict__ BT,
    float* __restrict__ C, int M, int N, int K) {
  const int tid  = threadIdx.x;
  const int lane = tid & 31;
  const int w    = tid >> 5;
  const int mi   = w & 1;            // 2 M sub-tiles of 32
  const int ni   = w >> 1;           // 4 N sub-tiles of 32
  const int m0   = blockIdx.y * 64;
  const int n0   = blockIdx.x * 128;
  const int lm   = lane & 15;
  const int hi   = lane >> 4;

  const _Float16* pa0 = A  + (size_t)(m0 + mi * 32 + lm)      * K;
  const _Float16* pa1 = A  + (size_t)(m0 + mi * 32 + 16 + lm) * K;
  const _Float16* pb0 = BT + (size_t)(n0 + ni * 32 + lm)      * K;
  const _Float16* pb1 = BT + (size_t)(n0 + ni * 32 + 16 + lm) * K;

  v8f acc00 = {}, acc01 = {}, acc10 = {}, acc11 = {};

  for (int kk = 0; kk < K; kk += 32) {
    const _Float16* a0p = pa0 + kk + hi * 16;
    const _Float16* a1p = pa1 + kk + hi * 16;
    v16h a0 = __builtin_shufflevector(*(const v8h*)a0p, *(const v8h*)(a0p + 8),
        0,1,2,3,4,5,6,7,8,9,10,11,12,13,14,15);
    v16h a1 = __builtin_shufflevector(*(const v8h*)a1p, *(const v8h*)(a1p + 8),
        0,1,2,3,4,5,6,7,8,9,10,11,12,13,14,15);
    v16h b0 = *(const v16h*)(pb0 + kk + hi * 16);
    v16h b1 = *(const v16h*)(pb1 + kk + hi * 16);

    acc00 = __builtin_amdgcn_wmma_f32_16x16x32_f16(
        false, a0, false, b0, (short)0, acc00, false, false);
    acc01 = __builtin_amdgcn_wmma_f32_16x16x32_f16(
        false, a0, false, b1, (short)0, acc01, false, false);
    acc10 = __builtin_amdgcn_wmma_f32_16x16x32_f16(
        false, a1, false, b0, (short)0, acc10, false, false);
    acc11 = __builtin_amdgcn_wmma_f32_16x16x32_f16(
        false, a1, false, b1, (short)0, acc11, false, false);

    // hide streamed-A latency (weights are L2-resident; A rows stream once)
    __builtin_prefetch(a0p + 256, 0, 3);
    __builtin_prefetch(a1p + 256, 0, 3);
  }

  const int mB0 = m0 + mi * 32 + hi * 8;
  const int mB1 = mB0 + 16;
  const int nC0 = n0 + ni * 32 + lm;
  const int nC1 = nC0 + 16;
  #pragma unroll
  for (int r = 0; r < 8; ++r) {
    C[(size_t)(mB0 + r) * N + nC0] = acc00[r];
    C[(size_t)(mB0 + r) * N + nC1] = acc01[r];
    C[(size_t)(mB1 + r) * N + nC0] = acc10[r];
    C[(size_t)(mB1 + r) * N + nC1] = acc11[r];
  }
}

// ---------------------------------------------------------------------------
// beta = sigmoid(x@Wb), g = -exp(A_log)*softplus(x@Wa + dt_bias). N=6 is far
// too narrow for WMMA; one thread per row, fp32.
// ---------------------------------------------------------------------------
__global__ void gdn_beta_g(const float* __restrict__ x,
                           const float* __restrict__ Wb,
                           const float* __restrict__ Wa,
                           const float* __restrict__ A_log,
                           const float* __restrict__ dt_bias,
                           float* __restrict__ beta, float* __restrict__ g) {
  int row = blockIdx.x * blockDim.x + threadIdx.x;
  if (row >= ROWS) return;
  float accB[H_] = {}, accA[H_] = {};
  const float* xr = x + (size_t)row * D_;
  for (int d = 0; d < D_; ++d) {
    float xd = xr[d];
    #pragma unroll
    for (int h = 0; h < H_; ++h) {
      accB[h] += xd * Wb[d * H_ + h];
      accA[h] += xd * Wa[d * H_ + h];
    }
  }
  #pragma unroll
  for (int h = 0; h < H_; ++h) {
    beta[row * H_ + h] = 1.f / (1.f + expf(-accB[h]));
    float z  = accA[h] + dt_bias[h];
    float sp = (z > 20.f) ? z : log1pf(expf(z));
    g[row * H_ + h] = -expf(A_log[h]) * sp;
  }
}

// ---------------------------------------------------------------------------
// Depthwise causal conv (KS=4) + SiLU, channels-last [B,L,C].
// ---------------------------------------------------------------------------
__global__ void gdn_conv_silu(const float* __restrict__ in,
                              const float* __restrict__ wc,
                              float* __restrict__ out, int C) {
  int idx = blockIdx.x * blockDim.x + threadIdx.x;
  if (idx >= ROWS * C) return;
  int c  = idx % C;
  int rl = idx / C;
  int l  = rl % L_;
  int b  = rl / L_;
  float acc = 0.f;
  #pragma unroll
  for (int j = 0; j < KS_; ++j) {
    int li = l - (KS_ - 1) + j;
    if (li >= 0)
      acc += in[(size_t)(b * L_ + li) * C + c] * wc[c * KS_ + j];
  }
  out[idx] = acc / (1.f + expf(-acc));   // SiLU
}

// ---------------------------------------------------------------------------
// Per-(row,head) l2 normalization over DK with post-scale (q: DK^-0.5, k: 1).
// ---------------------------------------------------------------------------
__global__ void gdn_l2norm(float* __restrict__ t, float post_scale) {
  int idx = blockIdx.x * blockDim.x + threadIdx.x;   // row*H + h
  if (idx >= ROWS * H_) return;
  int h = idx % H_, row = idx / H_;
  float* p = t + (size_t)row * KD_ + h * DK_;
  float ss = 1e-12f;
  for (int i = 0; i < DK_; ++i) { float v = p[i]; ss += v * v; }
  float sc = rsqrtf(ss) * post_scale;
  for (int i = 0; i < DK_; ++i) p[i] *= sc;
}

// ---------------------------------------------------------------------------
// Gated delta-rule scan. S columns evolve independently -> B*H*DV = 3072
// independent DK-deep recurrences. Block = 256 threads: one (b,h) x 64
// v-columns; 4 consecutive lanes share a column, 32 state floats each in
// VGPRs; quad reductions via __shfl_xor. Per-step k_t/q_t staging goes
// memory->LDS via GLOBAL_LOAD_ASYNC_TO_LDS_B32 (ASYNCcnt, no VGPR round
// trip); dynamic LDS so byte offsets are statically 0-based.
// LDS map: [0,512) k_t, [512,1024) q_t, [1024..] {exp(g), beta}.
// ---------------------------------------------------------------------------
__global__ __launch_bounds__(256) void gdn_scan(
    const float* __restrict__ q, const float* __restrict__ k,
    const float* __restrict__ v, const float* __restrict__ g,
    const float* __restrict__ beta, float* __restrict__ o,
    float* __restrict__ Sout) {
  extern __shared__ float smem[];   // 258 floats

  int bid = blockIdx.x;
  int vb  = bid & 3;               // DV block (4 x 64 columns)
  int h   = (bid >> 2) % H_;
  int b   = bid / (4 * H_);
  int t   = threadIdx.x;
  int col   = vb * 64 + (t >> 2);
  int sub   = t & 3;
  int kbase = sub * 32;

  float s[32];
  #pragma unroll
  for (int j = 0; j < 32; ++j) s[j] = 0.f;

  for (int l = 0; l < L_; ++l) {
    int row = b * L_ + l;
    // ---- async stage k_t (threads 0..127) / q_t (threads 128..255) to LDS
    if (t < DK_) {
      const float* src = k + (size_t)row * KD_ + h * DK_ + t;
      unsigned ldsoff = t * 4u;
      asm volatile("global_load_async_to_lds_b32 %0, %1, off"
                   :: "v"(ldsoff), "v"(src) : "memory");
    } else {
      const float* src = q + (size_t)row * KD_ + h * DK_ + (t - DK_);
      unsigned ldsoff = 512u + (t - DK_) * 4u;
      asm volatile("global_load_async_to_lds_b32 %0, %1, off"
                   :: "v"(ldsoff), "v"(src) : "memory");
    }
    if (t == 0) {
      smem[256] = expf(g[row * H_ + h]);
      smem[257] = beta[row * H_ + h];
    }
    asm volatile("s_wait_asynccnt 0x0" ::: "memory");
    __syncthreads();

    float eg = smem[256], bt = smem[257];
    float vt = v[(size_t)row * VD_ + h * DV_ + col];

    const float4* lk4 = (const float4*)&smem[kbase];
    const float4* lq4 = (const float4*)&smem[DK_ + kbase];
    float kl[32];
    float corr = 0.f;
    #pragma unroll
    for (int j4 = 0; j4 < 8; ++j4) {
      float4 kv = lk4[j4];
      kl[j4*4+0] = kv.x; kl[j4*4+1] = kv.y; kl[j4*4+2] = kv.z; kl[j4*4+3] = kv.w;
      corr += kv.x * s[j4*4+0] + kv.y * s[j4*4+1]
            + kv.z * s[j4*4+2] + kv.w * s[j4*4+3];
    }
    corr += __shfl_xor(corr, 1);
    corr += __shfl_xor(corr, 2);
    float u = (vt - eg * corr) * bt;   // corr taken on decayed state

    float oacc = 0.f;
    #pragma unroll
    for (int j4 = 0; j4 < 8; ++j4) {
      float4 qv = lq4[j4];
      float s0 = s[j4*4+0] * eg + kl[j4*4+0] * u; s[j4*4+0] = s0; oacc += qv.x * s0;
      float s1 = s[j4*4+1] * eg + kl[j4*4+1] * u; s[j4*4+1] = s1; oacc += qv.y * s1;
      float s2 = s[j4*4+2] * eg + kl[j4*4+2] * u; s[j4*4+2] = s2; oacc += qv.z * s2;
      float s3 = s[j4*4+3] * eg + kl[j4*4+3] * u; s[j4*4+3] = s3; oacc += qv.w * s3;
    }
    oacc += __shfl_xor(oacc, 1);
    oacc += __shfl_xor(oacc, 2);
    if (sub == 0) o[(size_t)row * VD_ + h * DV_ + col] = oacc;
    __syncthreads();
  }

  // final state S[b,h,k,v] (second tuple output)
  size_t sbase = (size_t)(b * H_ + h) * DK_ * DV_;
  #pragma unroll
  for (int j = 0; j < 32; ++j)
    Sout[sbase + (size_t)(kbase + j) * DV_ + col] = s[j];
}

// ---------------------------------------------------------------------------
// RMS-norm over DV + norm_w + SiLU(gate), emit f16 rows for final GEMM.
// ---------------------------------------------------------------------------
__global__ void gdn_normgate(const float* __restrict__ o,
                             const float* __restrict__ gate,
                             const float* __restrict__ norm_w,
                             _Float16* __restrict__ out16) {
  int idx = blockIdx.x * blockDim.x + threadIdx.x;   // row*H + h
  if (idx >= ROWS * H_) return;
  int h = idx % H_, row = idx / H_;
  const float* po = o    + (size_t)row * VD_ + h * DV_;
  const float* pg = gate + (size_t)row * VD_ + h * DV_;
  _Float16*    pd = out16 + (size_t)row * VD_ + h * DV_;
  float ss = 0.f;
  for (int i = 0; i < DV_; ++i) { float vv = po[i]; ss += vv * vv; }
  float inv = rsqrtf(ss / (float)DV_ + 1e-5f);
  for (int i = 0; i < DV_; ++i) {
    float gt  = pg[i];
    float sig = gt / (1.f + expf(-gt));
    pd[i] = (_Float16)(po[i] * inv * norm_w[i] * sig);
  }
}

// ---------------------------------------------------------------------------
// Launch
// ---------------------------------------------------------------------------
extern "C" void kernel_launch(void* const* d_in, const int* in_sizes, int n_in,
                              void* d_out, int out_size, void* d_ws, size_t ws_size,
                              hipStream_t stream) {
  const float* x       = (const float*)d_in[0];
  const float* Wq      = (const float*)d_in[1];
  const float* Wk      = (const float*)d_in[2];
  const float* Wv      = (const float*)d_in[3];
  const float* Wb      = (const float*)d_in[4];
  const float* Wa      = (const float*)d_in[5];
  const float* A_log   = (const float*)d_in[6];
  const float* dt_bias = (const float*)d_in[7];
  const float* conv_q  = (const float*)d_in[8];
  const float* conv_k  = (const float*)d_in[9];
  const float* conv_v  = (const float*)d_in[10];
  const float* Wg      = (const float*)d_in[11];
  const float* norm_w  = (const float*)d_in[12];
  const float* Wo      = (const float*)d_in[13];

  char* p = (char*)d_ws;
  auto carve = [&](size_t bytes) -> char* {
    char* r = p; p += (bytes + 255) & ~(size_t)255; return r;
  };
  _Float16* x16   = (_Float16*)carve((size_t)ROWS * D_  * 2);
  _Float16* WqT16 = (_Float16*)carve((size_t)D_   * KD_ * 2);  // [KD][D]
  _Float16* WkT16 = (_Float16*)carve((size_t)D_   * KD_ * 2);
  _Float16* WvT16 = (_Float16*)carve((size_t)D_   * VD_ * 2);  // [VD][D]
  _Float16* WgT16 = (_Float16*)carve((size_t)D_   * VD_ * 2);
  _Float16* WoT16 = (_Float16*)carve((size_t)VD_  * D_  * 2);  // [D][VD]
  float*    qpre  = (float*)carve((size_t)ROWS * KD_ * 4);
  float*    kpre  = (float*)carve((size_t)ROWS * KD_ * 4);
  float*    vpre  = (float*)carve((size_t)ROWS * VD_ * 4);
  float*    qc    = (float*)carve((size_t)ROWS * KD_ * 4);
  float*    kc    = (float*)carve((size_t)ROWS * KD_ * 4);
  float*    vc    = (float*)carve((size_t)ROWS * VD_ * 4);
  float*    gatef = (float*)carve((size_t)ROWS * VD_ * 4);
  float*    betab = (float*)carve((size_t)ROWS * H_  * 4);
  float*    gb    = (float*)carve((size_t)ROWS * H_  * 4);
  float*    obuf  = (float*)carve((size_t)ROWS * VD_ * 4);
  _Float16* of16  = (_Float16*)carve((size_t)ROWS * VD_ * 2);

  float* out_o = (float*)d_out;               // [B,L,D]
  float* out_S = out_o + (size_t)ROWS * D_;   // [B,H,DK,DV]

  gdn_cvt_f16<<<(ROWS * D_ + 255) / 256, 256, 0, stream>>>(x, x16, ROWS * D_);
  gdn_cvt_f16_t<<<(D_ * KD_ + 255) / 256, 256, 0, stream>>>(Wq, WqT16, D_, KD_);
  gdn_cvt_f16_t<<<(D_ * KD_ + 255) / 256, 256, 0, stream>>>(Wk, WkT16, D_, KD_);
  gdn_cvt_f16_t<<<(D_ * VD_ + 255) / 256, 256, 0, stream>>>(Wv, WvT16, D_, VD_);
  gdn_cvt_f16_t<<<(D_ * VD_ + 255) / 256, 256, 0, stream>>>(Wg, WgT16, D_, VD_);
  gdn_cvt_f16_t<<<(VD_ * D_ + 255) / 256, 256, 0, stream>>>(Wo, WoT16, VD_, D_);

  // Projections via LDS-free WMMA GEMMs (block tile 64x128)
  gdn_gemm_f16_wmma_nt<<<dim3(KD_ / 128, ROWS / 64), 256, 0, stream>>>(
      x16, WqT16, qpre, ROWS, KD_, D_);
  gdn_gemm_f16_wmma_nt<<<dim3(KD_ / 128, ROWS / 64), 256, 0, stream>>>(
      x16, WkT16, kpre, ROWS, KD_, D_);
  gdn_gemm_f16_wmma_nt<<<dim3(VD_ / 128, ROWS / 64), 256, 0, stream>>>(
      x16, WvT16, vpre, ROWS, VD_, D_);
  gdn_gemm_f16_wmma_nt<<<dim3(VD_ / 128, ROWS / 64), 256, 0, stream>>>(
      x16, WgT16, gatef, ROWS, VD_, D_);

  gdn_beta_g<<<(ROWS + 255) / 256, 256, 0, stream>>>(
      x, Wb, Wa, A_log, dt_bias, betab, gb);

  gdn_conv_silu<<<(ROWS * KD_ + 255) / 256, 256, 0, stream>>>(qpre, conv_q, qc, KD_);
  gdn_conv_silu<<<(ROWS * KD_ + 255) / 256, 256, 0, stream>>>(kpre, conv_k, kc, KD_);
  gdn_conv_silu<<<(ROWS * VD_ + 255) / 256, 256, 0, stream>>>(vpre, conv_v, vc, VD_);

  gdn_l2norm<<<(ROWS * H_ + 255) / 256, 256, 0, stream>>>(qc, 0.08838834764831845f);
  gdn_l2norm<<<(ROWS * H_ + 255) / 256, 256, 0, stream>>>(kc, 1.0f);

  gdn_scan<<<B_ * H_ * 4, 256, 258 * sizeof(float), stream>>>(
      qc, kc, vc, gb, betab, obuf, out_S);

  gdn_normgate<<<(ROWS * H_ + 255) / 256, 256, 0, stream>>>(obuf, gatef, norm_w, of16);

  gdn_gemm_f16_wmma_nt<<<dim3(D_ / 128, ROWS / 64), 256, 0, stream>>>(
      of16, WoT16, out_o, ROWS, D_, VD_);

  (void)in_sizes; (void)n_in; (void)out_size; (void)ws_size;
}
